// RevolutionaryPiMoE_2027224564542
// MI455X (gfx1250) — matrix operand order repaired and checked by reading, not compiled, not run on hardware
//
#include <hip/hip_runtime.h>
#include <stdint.h>

// ---------------------------------------------------------------------------
// Problem constants (match reference)
// ---------------------------------------------------------------------------
#define NTOK 2048
#define D_IN 1024
#define D_OUT 256
#define NEXP 50
#define HEXP 256
#define RH 512
#define ECAT (NEXP * D_OUT)   // 12800

// ---------------------------------------------------------------------------
// CDNA5 WMMA bf16 helpers (wave32, v_wmma_f32_16x16x32_bf16)
// ---------------------------------------------------------------------------
typedef __bf16 bf16x16 __attribute__((ext_vector_type(16)));
typedef float  f32x8   __attribute__((ext_vector_type(8)));
typedef unsigned int u32x4 __attribute__((ext_vector_type(4)));
typedef int          i32x4 __attribute__((ext_vector_type(4)));
typedef int          i32x8 __attribute__((ext_vector_type(8)));

union FragU {
    bf16x16        v;
    uint4          q[2];
    unsigned short u[16];
};

static __device__ __forceinline__ unsigned short f2bf(float f) {
    union { float f; uint32_t u; } x; x.f = f;
    uint32_t r = x.u + 0x7fffu + ((x.u >> 16) & 1u);   // round-to-nearest-even
    return (unsigned short)(r >> 16);
}
static __device__ __forceinline__ float bf2f(unsigned short b) {
    union { uint32_t u; float f; } x; x.u = ((uint32_t)b) << 16;
    return x.f;
}

// A fragment: 16(M) x 32(K) from row-major [*, lda] bf16, base already at (row0,k0).
// lane<16: row=lane,  K = {0..7, 16..23}; lane>=16: row=lane-16, K = {8..15, 24..31}.
static __device__ __forceinline__ FragU load_a(const unsigned short* A, int lda, int lane) {
    int half = lane >> 4, r = lane & 15;
    FragU f;
    const unsigned short* p = A + r * lda + half * 8;
    f.q[0] = *(const uint4*)(p);        // K = half*8 + 0..7
    f.q[1] = *(const uint4*)(p + 16);   // K = 16 + half*8 + 0..7
    return f;
}

// B fragment: 32(K) x 16(N) from row-major [K, ldb] bf16, base already at (k0,n0).
// lane<16: col=lane, K = {0..7, 16..23}; lane>=16: col=lane-16, K = {8..15, 24..31}.
static __device__ __forceinline__ FragU load_b(const unsigned short* B, int ldb, int lane) {
    FragU f;
    int half = lane >> 4, n = lane & 15;
    const unsigned short* p = B + n;
#pragma unroll
    for (int s = 0; s < 8; ++s) {
        f.u[s]     = p[(half * 8 + s) * ldb];
        f.u[8 + s] = p[(16 + half * 8 + s) * ldb];
    }
    return f;
}

static __device__ __forceinline__ f32x8 wmma_bf16(const FragU& a, const FragU& b, f32x8 c) {
    return __builtin_amdgcn_wmma_f32_16x16x32_bf16(false, a.v, false, b.v,
                                                   (short)0, c, false, false);
}

// ---------------------------------------------------------------------------
// f32 -> bf16 conversion
// ---------------------------------------------------------------------------
__global__ void cvt_bf16_kernel(const float* __restrict__ src,
                                unsigned short* __restrict__ dst, int n) {
    int i = blockIdx.x * blockDim.x + threadIdx.x;
    if (i < n) dst[i] = f2bf(src[i]);
}

// ---------------------------------------------------------------------------
// Fused per-expert MLP: cat[:, e*256:(e+1)*256] = relu(x@W1+b1) @ W2 + b2
// grid: (NTOK/16, NEXP), block: 256 (8 waves).
// The 16x1024 bf16 x-tile is DMA'd into LDS once per block via the Tensor
// Data Mover (TENSOR_LOAD_TO_LDS) with TDM padding: pad_interval=256 DW,
// pad_amount=4 DW -> LDS row stride 520 DW (bank-spread, 16B aligned).
// ---------------------------------------------------------------------------
#define XT_ROW_DW 520          // 512 data DWORDs + 2 pads of 4 DW per row

static __device__ __forceinline__ int xt_idx(int r, int kd) {
    // kd = DWORD index within the 512-DW logical row; pads after each 256 DW
    return r * XT_ROW_DW + kd + ((kd >> 8) << 2);
}

__global__ void expert_fused_kernel(const unsigned short* __restrict__ xb,   // [N, D_IN]
                                    const unsigned short* __restrict__ ew1,  // [E, D_IN, H]
                                    const float* __restrict__ eb1,           // [E, H]
                                    const unsigned short* __restrict__ ew2,  // [E, H, D_OUT]
                                    const float* __restrict__ eb2,           // [E, D_OUT]
                                    unsigned short* __restrict__ cat)        // [N, ECAT]
{
    __shared__ unsigned int   xt[16 * XT_ROW_DW];   // padded 16 x 1024 bf16 tile (~32.5 KB)
    __shared__ unsigned short hb[16 * HEXP];        // 16 x 256 bf16 hidden tile (8 KB)

    const int e    = blockIdx.y;
    const int row0 = blockIdx.x * 16;
    const int wave = threadIdx.x >> 5;
    const int lane = threadIdx.x & 31;
    const int half = lane >> 4, r = lane & 15;
    const int c0   = wave * 32;

    // ---- stage x tile into LDS ----
#if __has_builtin(__builtin_amdgcn_tensor_load_to_lds)
    if (threadIdx.x < 32) {   // one wave issues the TDM descriptor
        unsigned long long ga = (unsigned long long)(size_t)(xb + (size_t)row0 * D_IN);
        unsigned int lds_addr = (unsigned int)(size_t)(void*)xt;  // low 32b = LDS offset

        u32x4 g0;
        g0[0] = 1u;                                   // count=1, user mode, no gather
        g0[1] = lds_addr;                             // lds_addr
        g0[2] = (unsigned int)ga;                     // global_addr[31:0]
        g0[3] = (unsigned int)((ga >> 32) & 0x01FFFFFFu) | (2u << 30);  // addr[56:32] | type=2

        i32x8 g1;
        // wg_mask=0 | data_size=2 (4B) | pad_enable | pad_interval=7 (256DW) | pad_amount=3 (4DW)
        g1[0] = (int)((2u << 16) | (1u << 20) | (7u << 22) | (3u << 25));
        g1[1] = (int)(512u << 16);                    // tensor_dim0 = 512 DW (low 16)
        g1[2] = (int)((2048u & 0xFFFFu) << 16);       // tensor_dim1 = 2048 rows (low 16)
        g1[3] = (int)(512u << 16);                    // tile_dim0 = 512 DW
        g1[4] = 16;                                   // tile_dim1 = 16 rows, tile_dim2 = 0
        g1[5] = 512;                                  // tensor_dim0_stride = 512 DW (low 32)
        g1[6] = 0;                                    // stride0 hi / stride1 lo
        g1[7] = 0;                                    // stride1 hi
        i32x4 g2 = {0, 0, 0, 0}, g3 = {0, 0, 0, 0};  // unused (2D tensor)
        i32x8 g4 = {0, 0, 0, 0, 0, 0, 0, 0};         // unused trailing group (6-arg form)
        __builtin_amdgcn_tensor_load_to_lds(g0, g1, g2, g3, g4, 0);
#if __has_builtin(__builtin_amdgcn_s_wait_tensorcnt)
        __builtin_amdgcn_s_wait_tensorcnt(0);
#endif
    }
#else
    // fallback: cooperative copy with the same padded layout
    {
        const unsigned int* xw = (const unsigned int*)xb;
        for (int idx = threadIdx.x; idx < 16 * 512; idx += blockDim.x) {
            int rr = idx >> 9, kd = idx & 511;
            xt[xt_idx(rr, kd)] = xw[(size_t)(row0 + rr) * (D_IN / 2) + kd];
        }
    }
#endif
    __syncthreads();

    // ---- phase 1: h = relu(x @ W1 + b1), wave owns columns [c0, c0+32) ----
    {
        const unsigned short* W1 = ew1 + (size_t)e * D_IN * HEXP;
        f32x8 acc0 = {}, acc1 = {};
        for (int k = 0; k < D_IN; k += 32) {
            FragU a;                                   // A fragment from LDS
            int kd0 = (k >> 1) + half * 4;             // (k + half*8)/2 DWORDs
            a.q[0] = *(const uint4*)&xt[xt_idx(r, kd0)];
            a.q[1] = *(const uint4*)&xt[xt_idx(r, kd0 + 8)];
            FragU b0 = load_b(W1 + (size_t)k * HEXP + c0,      HEXP, lane);
            FragU b1 = load_b(W1 + (size_t)k * HEXP + c0 + 16, HEXP, lane);
            acc0 = wmma_bf16(a, b0, acc0);
            acc1 = wmma_bf16(a, b1, acc1);
        }
#pragma unroll
        for (int f = 0; f < 2; ++f) {
            f32x8& acc = f ? acc1 : acc0;
            int col = c0 + f * 16 + r;
            float bv = eb1[e * HEXP + col];
#pragma unroll
            for (int v = 0; v < 8; ++v) {
                int row = v + half * 8;
                hb[row * HEXP + col] = f2bf(fmaxf(acc[v] + bv, 0.0f));
            }
        }
    }
    __syncthreads();

    // ---- phase 2: out = h @ W2 + b2, wave owns output columns [c0, c0+32) ----
    {
        const unsigned short* W2 = ew2 + (size_t)e * HEXP * D_OUT;
        f32x8 acc0 = {}, acc1 = {};
        for (int k = 0; k < HEXP; k += 32) {
            FragU a;   // A fragment straight out of LDS (ds_read_b128)
            const unsigned short* p = &hb[r * HEXP + k + half * 8];
            a.q[0] = *(const uint4*)(p);
            a.q[1] = *(const uint4*)(p + 16);
            FragU b0 = load_b(W2 + (size_t)k * D_OUT + c0,      D_OUT, lane);
            FragU b1 = load_b(W2 + (size_t)k * D_OUT + c0 + 16, D_OUT, lane);
            acc0 = wmma_bf16(a, b0, acc0);
            acc1 = wmma_bf16(a, b1, acc1);
        }
#pragma unroll
        for (int f = 0; f < 2; ++f) {
            f32x8& acc = f ? acc1 : acc0;
            int col = c0 + f * 16 + r;
            float bv = eb2[e * D_OUT + col];
#pragma unroll
            for (int v = 0; v < 8; ++v) {
                int row = row0 + v + half * 8;
                cat[(size_t)row * ECAT + e * D_OUT + col] = f2bf(acc[v] + bv);
            }
        }
    }
}

// ---------------------------------------------------------------------------
// Generic bf16 WMMA GEMM: out = epilogue(A[rows,K] @ B[K,M] + bias)
// One wave per 16x64 output tile (4 accumulators, A fragment reused 4x).
// GATED: f32 out scaled by g1[row]*g2[row]; else bf16 out with optional ReLU.
// ---------------------------------------------------------------------------
template <bool RELU, bool GATED>
__global__ void gemm_wmma_kernel(const unsigned short* __restrict__ A,
                                 const unsigned short* __restrict__ B,
                                 const float* __restrict__ bias,
                                 unsigned short* __restrict__ outb,
                                 float* __restrict__ outf,
                                 const float* __restrict__ g1,
                                 const float* __restrict__ g2,
                                 int K, int M, int tilesPerRow, int totalTiles)
{
    const int wave = threadIdx.x >> 5;
    const int lane = threadIdx.x & 31;
    const int tid  = blockIdx.x * (blockDim.x >> 5) + wave;
    if (tid >= totalTiles) return;   // uniform per wave

    const int row0 = (tid / tilesPerRow) * 16;
    const int col0 = (tid % tilesPerRow) * 64;
    const int half = lane >> 4, r = lane & 15;

    f32x8 acc[4] = {};
    for (int k = 0; k < K; k += 32) {
        // prefetch the streamed B operand 8 K-steps ahead (global_prefetch_b8)
        if (k + 256 < K)
            __builtin_prefetch(B + (size_t)(k + 256) * M + col0 + lane * M, 0, 0);
        FragU a = load_a(A + (size_t)row0 * K + k, K, lane);
#pragma unroll
        for (int f = 0; f < 4; ++f) {
            FragU b = load_b(B + (size_t)k * M + col0 + f * 16, M, lane);
            acc[f] = wmma_bf16(a, b, acc[f]);
        }
    }
#pragma unroll
    for (int f = 0; f < 4; ++f) {
        int col = col0 + f * 16 + r;
        float bv = bias[col];
#pragma unroll
        for (int v = 0; v < 8; ++v) {
            int row = row0 + v + half * 8;
            float val = acc[f][v] + bv;
            if (RELU) val = fmaxf(val, 0.0f);
            if (GATED) outf[(size_t)row * M + col] = val * g1[row] * g2[row];
            else       outb[(size_t)row * M + col] = f2bf(val);
        }
    }
}

// ---------------------------------------------------------------------------
// sigmoid(H[row,:256] . w + b) per row; one wave per row (wave32 shfl reduce)
// ---------------------------------------------------------------------------
__global__ void gemv_sigmoid_kernel(const unsigned short* __restrict__ Hb, // [N,256] bf16
                                    const float* __restrict__ w,           // [256]
                                    const float* __restrict__ b,           // [1]
                                    float* __restrict__ out)               // [N]
{
    const int wave = threadIdx.x >> 5;
    const int lane = threadIdx.x & 31;
    const int row  = blockIdx.x * (blockDim.x >> 5) + wave;
    float s = 0.0f;
#pragma unroll
    for (int j = lane; j < 256; j += 32)
        s += bf2f(Hb[(size_t)row * 256 + j]) * w[j];
#pragma unroll
    for (int off = 16; off > 0; off >>= 1)
        s += __shfl_xor(s, off, 32);
    if (lane == 0) out[row] = 1.0f / (1.0f + __expf(-(s + b[0])));
}

// ---------------------------------------------------------------------------
// Host launcher
// ---------------------------------------------------------------------------
extern "C" void kernel_launch(void* const* d_in, const int* in_sizes, int n_in,
                              void* d_out, int out_size, void* d_ws, size_t ws_size,
                              hipStream_t stream) {
    (void)in_sizes; (void)n_in; (void)out_size; (void)ws_size;

    const float* x   = (const float*)d_in[0];
    const float* ew1 = (const float*)d_in[1];
    const float* eb1 = (const float*)d_in[2];
    const float* ew2 = (const float*)d_in[3];
    const float* eb2 = (const float*)d_in[4];
    const float* dw1 = (const float*)d_in[5];
    const float* db1 = (const float*)d_in[6];
    const float* dw2 = (const float*)d_in[7];
    const float* db2 = (const float*)d_in[8];
    const float* lw1 = (const float*)d_in[9];
    const float* lb1 = (const float*)d_in[10];
    const float* lw2 = (const float*)d_in[11];
    const float* lb2 = (const float*)d_in[12];
    const float* bw1 = (const float*)d_in[13];
    const float* bb1 = (const float*)d_in[14];
    const float* bw2 = (const float*)d_in[15];
    const float* bb2 = (const float*)d_in[16];
    const float* pw1 = (const float*)d_in[17];
    const float* pb1 = (const float*)d_in[18];
    const float* pw2 = (const float*)d_in[19];
    const float* pb2 = (const float*)d_in[20];
    float* out = (float*)d_out;

    // ---- workspace bump allocator (256 B aligned) ----
    char* wp = (char*)d_ws;
    auto alloc = [&](size_t bytes) -> char* {
        char* r = wp;
        wp += (bytes + 255) & ~(size_t)255;
        return r;
    };
    unsigned short* xb    = (unsigned short*)alloc((size_t)NTOK * D_IN * 2);
    unsigned short* ew1b  = (unsigned short*)alloc((size_t)NEXP * D_IN * HEXP * 2);
    unsigned short* ew2b  = (unsigned short*)alloc((size_t)NEXP * HEXP * D_OUT * 2);
    unsigned short* dw1b  = (unsigned short*)alloc((size_t)ECAT * RH * 2);
    unsigned short* dw2b  = (unsigned short*)alloc((size_t)RH * D_OUT * 2);
    unsigned short* lw1b  = (unsigned short*)alloc((size_t)D_IN * 256 * 2);
    unsigned short* bw1b  = (unsigned short*)alloc((size_t)D_IN * 256 * 2);
    unsigned short* pw1b  = (unsigned short*)alloc((size_t)D_OUT * 256 * 2);
    unsigned short* pw2b  = (unsigned short*)alloc((size_t)256 * D_OUT * 2);
    unsigned short* cat   = (unsigned short*)alloc((size_t)NTOK * ECAT * 2);
    unsigned short* disth = (unsigned short*)alloc((size_t)NTOK * RH * 2);
    unsigned short* distb = (unsigned short*)alloc((size_t)NTOK * D_OUT * 2);
    unsigned short* lath  = (unsigned short*)alloc((size_t)NTOK * 256 * 2);
    unsigned short* bwdh  = (unsigned short*)alloc((size_t)NTOK * 256 * 2);
    unsigned short* privh = (unsigned short*)alloc((size_t)NTOK * 256 * 2);
    float*          latv  = (float*)alloc((size_t)NTOK * 4);
    float*          bwdv  = (float*)alloc((size_t)NTOK * 4);

    auto cvt = [&](const float* s, unsigned short* d, int n) {
        cvt_bf16_kernel<<<(n + 255) / 256, 256, 0, stream>>>(s, d, n);
    };
    cvt(x,   xb,   NTOK * D_IN);
    cvt(ew1, ew1b, NEXP * D_IN * HEXP);
    cvt(ew2, ew2b, NEXP * HEXP * D_OUT);
    cvt(dw1, dw1b, ECAT * RH);
    cvt(dw2, dw2b, RH * D_OUT);
    cvt(lw1, lw1b, D_IN * 256);
    cvt(bw1, bw1b, D_IN * 256);
    cvt(pw1, pw1b, D_OUT * 256);
    cvt(pw2, pw2b, 256 * D_OUT);

    // ---- fused expert MLPs -> cat ----
    expert_fused_kernel<<<dim3(NTOK / 16, NEXP), 256, 0, stream>>>(
        xb, ew1b, eb1, ew2b, eb2, cat);

    // ---- distributed_processor FC1: relu(cat @ dw1 + db1) -> disth ----
    {
        int tpr = RH / 64, tiles = (NTOK / 16) * tpr;
        gemm_wmma_kernel<true, false><<<(tiles + 7) / 8, 256, 0, stream>>>(
            cat, dw1b, db1, disth, nullptr, nullptr, nullptr, ECAT, RH, tpr, tiles);
    }
    // ---- distributed_processor FC2: disth @ dw2 + db2 -> distb ----
    {
        int tpr = D_OUT / 64, tiles = (NTOK / 16) * tpr;
        gemm_wmma_kernel<false, false><<<(tiles + 7) / 8, 256, 0, stream>>>(
            disth, dw2b, db2, distb, nullptr, nullptr, nullptr, RH, D_OUT, tpr, tiles);
    }
    // ---- gate hidden layers: relu(x @ {lw1,bw1} + b) ----
    {
        int tpr = 256 / 64, tiles = (NTOK / 16) * tpr;
        gemm_wmma_kernel<true, false><<<(tiles + 7) / 8, 256, 0, stream>>>(
            xb, lw1b, lb1, lath, nullptr, nullptr, nullptr, D_IN, 256, tpr, tiles);
        gemm_wmma_kernel<true, false><<<(tiles + 7) / 8, 256, 0, stream>>>(
            xb, bw1b, bb1, bwdh, nullptr, nullptr, nullptr, D_IN, 256, tpr, tiles);
    }
    // ---- gate scalars: sigmoid(h . w2 + b2) ----
    gemv_sigmoid_kernel<<<NTOK / 8, 256, 0, stream>>>(lath, lw2, lb2, latv);
    gemv_sigmoid_kernel<<<NTOK / 8, 256, 0, stream>>>(bwdh, bw2, bb2, bwdv);

    // ---- privacy_preserver FC1: relu(dist @ pw1 + pb1) ----
    {
        int tpr = 256 / 64, tiles = (NTOK / 16) * tpr;
        gemm_wmma_kernel<true, false><<<(tiles + 7) / 8, 256, 0, stream>>>(
            distb, pw1b, pb1, privh, nullptr, nullptr, nullptr, D_OUT, 256, tpr, tiles);
        // ---- FC2 + fused lat*bwd gating -> f32 d_out ----
        gemm_wmma_kernel<false, true><<<(tiles + 7) / 8, 256, 0, stream>>>(
            privh, pw2b, pb2, nullptr, out, latv, bwdv, 256, D_OUT, tpr, tiles);
    }
}
